// LatentBKI_29506425323753
// MI455X (gfx1250) — matrix operand order repaired and checked by reading, not compiled, not run on hardware
//
#include <hip/hip_runtime.h>
#include <math.h>

typedef __attribute__((ext_vector_type(16))) _Float16 v16h;
typedef __attribute__((ext_vector_type(8)))  float    v8f;

#define GXD 128
#define GYD 128
#define GZD 32
#define GTOT (GXD*GYD*GZD)   // 524288
#define DD   64
#define ROW  67              // xyz + 64 feats
#define MAXL 512             // cap on contributors per 16-cell tile (avg ~31)

struct W8 { float w[8]; };   // weight LUT indexed |dx|*4 + |dy|*2 + |dz|

struct __align__(16) U128 { unsigned a, b, c, d; };

__device__ __forceinline__ int cell_of(float px, float py, float pz) {
    int ix = (int)floorf((px + 20.0f) / 0.3125f);
    int iy = (int)floorf((py + 20.0f) / 0.3125f);
    int iz = (int)floorf((pz +  2.0f) / 0.25f);
    ix = min(max(ix, 0), GXD - 1);
    iy = min(max(iy, 0), GYD - 1);
    iz = min(max(iz, 0), GZD - 1);
    return (ix * GYD + iy) * GZD + iz;
}

// ---------------- binning: count -> scan -> scatter -> per-bin sort ----------

__global__ void k_count(const float* __restrict__ pc, int* __restrict__ counts, int N) {
    int i = blockIdx.x * blockDim.x + threadIdx.x;
    if (i < N) {
        const float* p = pc + (long long)i * ROW;
        atomicAdd(&counts[cell_of(p[0], p[1], p[2])], 1);
    }
}

__global__ __launch_bounds__(256) void k_scan1(const int* __restrict__ counts,
                                               int* __restrict__ offs,
                                               int* __restrict__ bsums) {
    __shared__ int sh[256];
    int t = threadIdx.x, b = blockIdx.x;
    int base = b * 1024 + t * 4;
    int c0 = counts[base + 0], c1 = counts[base + 1];
    int c2 = counts[base + 2], c3 = counts[base + 3];
    int tot = c0 + c1 + c2 + c3;
    sh[t] = tot; __syncthreads();
    for (int o = 1; o < 256; o <<= 1) {
        int v = (t >= o) ? sh[t - o] : 0;
        __syncthreads();
        sh[t] += v;
        __syncthreads();
    }
    int ex = sh[t] - tot;
    offs[base + 0] = ex;
    offs[base + 1] = ex + c0;
    offs[base + 2] = ex + c0 + c1;
    offs[base + 3] = ex + c0 + c1 + c2;
    if (t == 255) bsums[b] = sh[255];
}

__global__ __launch_bounds__(512) void k_scan2(int* __restrict__ bsums) {
    __shared__ int sh[512];
    int t = threadIdx.x;
    int v0 = bsums[t];
    sh[t] = v0; __syncthreads();
    for (int o = 1; o < 512; o <<= 1) {
        int v = (t >= o) ? sh[t - o] : 0;
        __syncthreads();
        sh[t] += v;
        __syncthreads();
    }
    bsums[t] = sh[t] - v0;   // exclusive
}

__global__ void k_scan3(int* __restrict__ offs, const int* __restrict__ bsums,
                        int* __restrict__ cursors) {
    int i = blockIdx.x * blockDim.x + threadIdx.x;
    if (i < GTOT) {
        int v = offs[i] + bsums[i >> 10];
        offs[i] = v;
        cursors[i] = v;
    }
}

__global__ void k_scatter(const float* __restrict__ pc, int* __restrict__ cursors,
                          int* __restrict__ binned, int N) {
    int i = blockIdx.x * blockDim.x + threadIdx.x;
    if (i < N) {
        const float* p = pc + (long long)i * ROW;
        int c = cell_of(p[0], p[1], p[2]);
        int s = atomicAdd(&cursors[c], 1);
        binned[s] = i;
    }
}

// Canonicalize per-bin order so FP accumulation order is run-to-run deterministic.
__global__ void k_sort(const int* __restrict__ counts, const int* __restrict__ offs,
                       int* __restrict__ binned) {
    int c = blockIdx.x * blockDim.x + threadIdx.x;
    if (c < GTOT) {
        int len = counts[c];
        if (len > 1) {
            int s = offs[c];
            for (int i = 1; i < len; i++) {
                int v = binned[s + i];
                int j = i - 1;
                while (j >= 0 && binned[s + j] > v) { binned[s + j + 1] = binned[s + j]; j--; }
                binned[s + j + 1] = v;
            }
        }
    }
}

// ---------------- fused gather + WMMA accumulate + Bayesian update -----------
// One block = 16 cells (fixed x,y; z0..z0+15). 4 waves, wave w owns features
// [16w, 16w+16). Per K-chunk of 32 contributor points:
//   A (16x32 f16)  = kernel weights per (cell, point)
//   By (32x16 f16) = features; feature^2 operand is computed in-register
//   (bsv = byv*byv, 8x v_pk_mul_f16) -> two chained v_wmma per wave.

__global__ __launch_bounds__(128) void bki_main(
    const float* __restrict__ mean_map, const float* __restrict__ var_map,
    const float* __restrict__ conf_map, const float* __restrict__ pc,
    const int* __restrict__ counts, const int* __restrict__ offsets,
    const int* __restrict__ binned,
    float* __restrict__ out_mean, float* __restrict__ out_var,
    float* __restrict__ out_conf, W8 wk)
{
    const int t    = threadIdx.x;
    const int tile = blockIdx.x;
    const int z0   = (tile & 1) * 16;
    const int xy   = tile >> 1;
    const int y    = xy & (GYD - 1);
    const int x    = xy >> 7;
    const long long cellbase = ((long long)x * GYD + y) * GZD + z0;

    __shared__ int scCnt[162], scStart[162], scPos[162];
    __shared__ int Lsh;
    __shared__ unsigned plist[MAXL];                 // pid | (meta<<20)
    __shared__ __align__(32) _Float16 Alds[512];     // 16x32, lane-swizzled
    __shared__ __align__(32) _Float16 By[2048];      // 4 waves x 32x16
    __shared__ float kbar[16], confl[16], w8s[8];

    if (t < 8)  w8s[t] = wk.w[t];
    if (t < 16) { kbar[t] = 0.0f; confl[t] = conf_map[cellbase + t]; }

    // enumerate the 9 x 18 source cells of this tile's neighborhood
    for (int s = t; s < 162; s += 128) {
        int dxy = s / 18, zz = s % 18;
        int dx = dxy / 3 - 1, dy = dxy % 3 - 1;
        int bx = x + dx, by = y + dy, bz = z0 - 1 + zz;
        int cnt = 0, st = 0;
        if (bx >= 0 && bx < GXD && by >= 0 && by < GYD && bz >= 0 && bz < GZD) {
            int c = (bx * GYD + by) * GZD + bz;
            cnt = counts[c]; st = offsets[c];
        }
        scCnt[s] = cnt; scStart[s] = st;
    }
    __syncthreads();
    if (t == 0) {                      // deterministic serial scan (162 elems)
        int acc = 0;
        for (int s = 0; s < 162; s++) { scPos[s] = acc; acc += scCnt[s]; }
        Lsh = (acc < MAXL) ? acc : MAXL;
    }
    __syncthreads();
    for (int s = t; s < 162; s += 128) {
        int cnt = scCnt[s], pos = scPos[s], st = scStart[s];
        int dxy = s / 18, zz = s % 18;
        int dx = dxy / 3 - 1, dy = dxy % 3 - 1;
        unsigned dxa = (unsigned)(dx < 0 ? -dx : dx);
        unsigned dya = (unsigned)(dy < 0 ? -dy : dy);
        unsigned meta = dxa | (dya << 1) | ((unsigned)zz << 2);
        for (int i = 0; i < cnt; i++) {
            int p = pos + i;
            if (p < MAXL) plist[p] = (unsigned)binned[st + i] | (meta << 20);
        }
    }
    __syncthreads();
    const int L    = Lsh;
    const int w    = t >> 5;       // wave id (wave32)
    const int lane = t & 31;

    v8f cy = {};   // sum k*f   (16 cells x 16 feats)
    v8f cs = {};   // sum k*f^2

    for (int k0 = 0; k0 < L; k0 += 32) {
        // A full chunk rewrites every A/By address; zero only a partial chunk.
        const bool partial = (k0 + 32 > L);   // block-uniform
        if (partial) {
            U128 z; z.a = z.b = z.c = z.d = 0u;
            if (t < 64) ((U128*)Alds)[t] = z;                 // 512 halfs
            for (int i = t; i < 256; i += 128) ((U128*)By)[i] = z;  // 2048 halfs
            __syncthreads();
        }

        // stage features: 128 threads = 2 points x 64 feats per pass
        for (int kk = 0; kk < 32; kk += 2) {
            int slot = kk + (t >> 6);
            int k = k0 + slot;
            if (k < L) {
                unsigned e = plist[k];
                int pid = (int)(e & 0xFFFFFu);
                int d   = t & 63;
                float f = pc[(long long)pid * ROW + 3 + d];
                // B(32x16) lane layout: lane = 16*((slot>>3)&1) + n, elem j = (slot&7) + 8*(slot>>4)
                int addr = (((d >> 4) * 32) + 16 * ((slot >> 3) & 1) + (d & 15)) * 16 + ((slot & 7) + 8 * (slot >> 4));
                By[addr] = (_Float16)f;
            }
        }
        // weights -> A panel (pre-swizzled for per-lane contiguous v16h read)
        for (int i = t; i < 512; i += 128) {
            int slot = i >> 4, m = i & 15;
            int k = k0 + slot;
            if (k < L) {
                unsigned e = plist[k];
                unsigned meta = e >> 20;
                int dxa = (int)(meta & 1), dya = (int)((meta >> 1) & 1);
                int zz  = (int)(meta >> 2);
                int dz  = zz - 1 - m;
                int dza = dz < 0 ? -dz : dz;
                float wgt = (dza <= 1) ? w8s[dxa * 4 + dya * 2 + dza] : 0.0f;
                int hia = (slot >> 3) & 1;
                int ja  = (slot & 7) + 8 * (slot >> 4);
                Alds[(m + 16 * hia) * 16 + ja] = (_Float16)wgt;
            }
        }
        __syncthreads();

        if (t < 16) {   // k_bar row sums: single owner -> deterministic
            v16h ra = *(const v16h*)&Alds[t * 16];
            v16h rb = *(const v16h*)&Alds[(t + 16) * 16];
            float s = 0.0f;
            #pragma unroll
            for (int j = 0; j < 16; j++) s += (float)ra[j] + (float)rb[j];
            kbar[t] += s;
        }

        v16h a   = *(const v16h*)&Alds[lane * 16];
        v16h byv = *(const v16h*)&By[(w * 32 + lane) * 16];
        v16h bsv = byv * byv;   // f^2 operand in registers: 8x v_pk_mul_f16
        cy = __builtin_amdgcn_wmma_f32_16x16x32_f16(false, a, false, byv, (short)0, cy, false, false);
        cs = __builtin_amdgcn_wmma_f32_16x16x32_f16(false, a, false, bsv, (short)0, cs, false, false);
        __syncthreads();
    }
    __syncthreads();

    // C/D layout: lane n = lane&15, hi = lane>>4; vgpr r -> cell row r + 8*hi
    const int n  = lane & 15;
    const int hi = lane >> 4;
    #pragma unroll
    for (int r = 0; r < 8; r++) {
        int m = r + 8 * hi;
        long long cell = cellbase + m;
        int fd = w * 16 + n;
        float kb   = kbar[m];
        float cf   = confl[m];
        float ysum = cy[r];
        float ybar = ysum / (kb + 1e-6f);
        float Sbar = cs[r] - ybar * (2.0f * ysum - ybar * kb);
        long long idx = cell * DD + fd;
        float mean = mean_map[idx];
        float var  = var_map[idx];
        float dm   = ybar - mean;
        float den  = cf + kb + 1e-6f;
        float scal = cf * kb / den;
        out_mean[idx] = (cf * mean + kb * ybar) / den;
        out_var[idx]  = var + Sbar + scal * dm * dm;
    }
    if (t < 16) out_conf[cellbase + t] = confl[t] + kbar[t];
}

// ---------------- host side ----------------

static float sparse_k_host(float d) {
    const float ell = 0.5f;
    const float pi  = 3.14159265358979323846f;
    if (d >= ell) return 0.0f;
    float v = (1.0f / 3.0f) * (2.0f + cosf(2.0f * pi * d / ell)) * (1.0f - d / ell)
            + (1.0f / (2.0f * pi)) * sinf(2.0f * pi * d / ell);
    if (v < 0.0f) v = 0.0f;
    if (v > 1.0f) v = 1.0f;
    return v;
}

extern "C" void kernel_launch(void* const* d_in, const int* in_sizes, int n_in,
                              void* d_out, int out_size, void* d_ws, size_t ws_size,
                              hipStream_t stream) {
    const float* mean_map = (const float*)d_in[0];
    const float* var_map  = (const float*)d_in[1];
    const float* conf_map = (const float*)d_in[2];
    const float* pc       = (const float*)d_in[3];
    const int N = in_sizes[3] / ROW;

    float* out_mean = (float*)d_out;
    float* out_var  = out_mean + (long long)GTOT * DD;
    float* out_conf = out_var  + (long long)GTOT * DD;

    int* counts  = (int*)d_ws;
    int* offs    = counts  + GTOT;
    int* cursors = offs    + GTOT;
    int* bsums   = cursors + GTOT;
    int* binned  = bsums   + 1024;

    // 8-entry weight LUT: offsets are ±1 voxel; VOX = (0.3125, 0.3125, 0.25)
    W8 wk;
    const float vx = 0.3125f, vy = 0.3125f, vz = 0.25f;
    for (int a = 0; a < 2; a++)
        for (int b = 0; b < 2; b++)
            for (int c = 0; c < 2; c++) {
                float d = sqrtf((float)a * vx * vx + (float)b * vy * vy + (float)c * vz * vz);
                wk.w[a * 4 + b * 2 + c] = sparse_k_host(d);
            }

    hipMemsetAsync(counts, 0, (size_t)GTOT * sizeof(int), stream);
    k_count  <<<(N + 255) / 256, 256, 0, stream>>>(pc, counts, N);
    k_scan1  <<<GTOT / 1024, 256, 0, stream>>>(counts, offs, bsums);
    k_scan2  <<<1, 512, 0, stream>>>(bsums);
    k_scan3  <<<GTOT / 256, 256, 0, stream>>>(offs, bsums, cursors);
    k_scatter<<<(N + 255) / 256, 256, 0, stream>>>(pc, cursors, binned, N);
    k_sort   <<<GTOT / 256, 256, 0, stream>>>(counts, offs, binned);
    bki_main <<<GXD * GYD * (GZD / 16), 128, 0, stream>>>(
        mean_map, var_map, conf_map, pc, counts, offs, binned,
        out_mean, out_var, out_conf, wk);
}
